// DifferentiableLogicMachine_15564961481116
// MI455X (gfx1250) — matrix operand order repaired and checked by reading, not compiled, not run on hardware
//
#include <hip/hip_runtime.h>
#include <hip/hip_bf16.h>

typedef _Float16 h8   __attribute__((ext_vector_type(8)));
typedef _Float16 v16h __attribute__((ext_vector_type(16)));
typedef float    v8f  __attribute__((ext_vector_type(8)));

union H16 { v16h v; h8 h[2]; };

#define B_  16
#define N_  128
#define C_  64

// ---------------- Kernel A: out2 GEMM + fused x2 reductions ----------------
// LDS half-strides padded for bank-conflict-free ds_load_b128 across 16 rows.
#define WT_STRIDE  264            // Wt[64][264]  (k-major weights, transposed)
#define ROW_STRIDE 72             // x-slabs [128][72]
#define OFF_WT     0
#define OFF_X1S    (64*WT_STRIDE)             // 16896 halves
#define OFF_XR     (OFF_X1S + 128*ROW_STRIDE) // 26112
#define OFF_XC     (OFF_XR  + 128*ROW_STRIDE) // 35328
#define HALVES_A   (OFF_XC  + 128*ROW_STRIDE) // 44544 halves
#define OFF_BIAS_A (HALVES_A*2)               // byte offset of float area
#define OFF_REDMAX (OFF_BIAS_A + 64*4)
#define OFF_REDMIN (OFF_REDMAX + 256*4)
#define SMEM_A_BYTES (OFF_REDMIN + 256*4)     // 91392 B (< 320KB WGP LDS)

__global__ __launch_bounds__(256)
void dlm_binary_kernel(const float* __restrict__ x1, const float* __restrict__ x2,
                       const float* __restrict__ W2, const float* __restrict__ b2,
                       float* __restrict__ out2,
                       float* __restrict__ rmaxW, float* __restrict__ rminW)
{
    extern __shared__ char smem[];
    _Float16* hs     = (_Float16*)smem;
    float*    biasL  = (float*)(smem + OFF_BIAS_A);
    float*    redmax = (float*)(smem + OFF_REDMAX);
    float*    redmin = (float*)(smem + OFF_REDMIN);

    const int t = threadIdx.x;
    const int b = blockIdx.x >> 7;      // / 128
    const int i = blockIdx.x & 127;

    // W2 [256][64] f32 -> Wt[n][k] f16 (k-major so B-frags are contiguous)
    #pragma unroll 4
    for (int e = 0; e < 64; ++e) {
        int idx = t + 256*e;
        int k = idx >> 6, n = idx & 63;
        hs[OFF_WT + n*WT_STRIDE + k] = (_Float16)W2[idx];
    }
    if (t < 64) biasL[t] = b2[t];

    // x1[b] slab [128][64] -> f16
    {
        const float* p = x1 + (size_t)b*N_*C_;
        int c = t & 63, jb = t >> 6;
        #pragma unroll 4
        for (int e = 0; e < 32; ++e) {
            int r = jb + 4*e;
            hs[OFF_X1S + r*ROW_STRIDE + c] = (_Float16)p[r*64 + c];
        }
    }

    // xr = x2[b][i][:][:] (contiguous 32KB) + fused masked max/min over j
    {
        const float* p = x2 + (((size_t)b*N_ + i)*N_)*C_;
        int c = t & 63, jb = t >> 6;       // each thread: fixed c, 32 j's
        float vmax = 0.0f, vmin = 1.0f;    // diagonal fill values (ref semantics)
        #pragma unroll 4
        for (int e = 0; e < 32; ++e) {
            int j = jb + 4*e;
            float v = p[j*64 + c];
            hs[OFF_XR + j*ROW_STRIDE + c] = (_Float16)v;
            vmax = fmaxf(vmax, (j == i) ? 0.0f : v);
            vmin = fminf(vmin, (j == i) ? 1.0f : v);
        }
        redmax[t] = vmax; redmin[t] = vmin;
    }

    // xc = x2[b][:][i][:] (transpose gather; L2-resident re-read)
    {
        int j = t >> 1, hoff = (t & 1)*32;
        const float4* p = (const float4*)(x2 + (((size_t)b*N_ + j)*N_ + i)*C_ + hoff);
        #pragma unroll
        for (int q = 0; q < 8; ++q) {
            float4 v = p[q];
            int base = OFF_XC + j*ROW_STRIDE + hoff + q*4;
            hs[base+0]=(_Float16)v.x; hs[base+1]=(_Float16)v.y;
            hs[base+2]=(_Float16)v.z; hs[base+3]=(_Float16)v.w;
        }
    }

    __syncthreads();

    // finish rmax/rmin (combine 4 partials per channel), spill to workspace
    if (t < 64) {
        float m = fmaxf(fmaxf(redmax[t], redmax[t+64]), fmaxf(redmax[t+128], redmax[t+192]));
        float n = fminf(fminf(redmin[t], redmin[t+64]), fminf(redmin[t+128], redmin[t+192]));
        size_t o = ((size_t)b*N_ + i)*C_ + t;
        rmaxW[o] = m; rminW[o] = n;
    }

    // ---- WMMA: each wave -> one 16(j) x 64(C) tile; K = 256 = 8 chunks ----
    const int wave = t >> 5, lane = t & 31;
    const int j0 = wave * 16;
    const int r = lane & 15, hi = lane >> 4;

    v8f acc[4] = {};

    #pragma unroll
    for (int kc = 0; kc < 8; ++kc) {
        // feature segments: [x1[b,i] | x2[b,i,j] | x1[b,j] | x2[b,j,i]]
        int seg = kc >> 1;
        int rowbase;
        if      (seg == 0) rowbase = OFF_X1S + i*ROW_STRIDE;          // broadcast
        else if (seg == 1) rowbase = OFF_XR  + (j0 + r)*ROW_STRIDE;
        else if (seg == 2) rowbase = OFF_X1S + (j0 + r)*ROW_STRIDE;
        else               rowbase = OFF_XC  + (j0 + r)*ROW_STRIDE;
        int off = rowbase + (kc & 1)*32 + hi*8;
        H16 a;
        a.h[0] = *(const h8*)(hs + off);        // K 0-7 / 8-15
        a.h[1] = *(const h8*)(hs + off + 16);   // K 16-23 / 24-31
        #pragma unroll
        for (int nb = 0; nb < 4; ++nb) {
            int n  = (lane & 15) + nb*16;
            int wo = OFF_WT + n*WT_STRIDE + kc*32 + hi*8;
            H16 bb;
            bb.h[0] = *(const h8*)(hs + wo);
            bb.h[1] = *(const h8*)(hs + wo + 16);
            acc[nb] = __builtin_amdgcn_wmma_f32_16x16x32_f16(
                false, a.v, false, bb.v, (short)0, acc[nb], false, false);
        }
    }

    // bias + sigmoid + store (D layout: VGPR v, lane<16 -> M=v, lane>=16 -> M=v+8)
    #pragma unroll
    for (int nb = 0; nb < 4; ++nb) {
        int col = (lane & 15) + nb*16;
        float bv = biasL[col];
        #pragma unroll
        for (int v = 0; v < 8; ++v) {
            int j = j0 + v + hi*8;
            float x = acc[nb][v] + bv;
            out2[(((size_t)b*N_ + i)*N_ + j)*C_ + col] = 1.0f/(1.0f + __expf(-x));
        }
    }
}

// ---------------- Kernel B: out1 GEMM (K=200 padded to 224) ----------------
#define BSTRIDE    232
#define OFF_A1     0
#define OFF_WT1    (128*BSTRIDE)             // 29696 halves
#define HALVES_B   (OFF_WT1 + 64*BSTRIDE)    // 44544
#define OFF_BIAS_B (HALVES_B*2)
#define SMEM_B_BYTES (OFF_BIAS_B + 64*4)     // 89344 B

__global__ __launch_bounds__(256)
void dlm_unary_kernel(const float* __restrict__ x0, const float* __restrict__ x1,
                      const float* __restrict__ W1, const float* __restrict__ b1,
                      const float* __restrict__ rmaxW, const float* __restrict__ rminW,
                      float* __restrict__ out1)
{
    extern __shared__ char smem[];
    _Float16* hs    = (_Float16*)smem;
    float*    biasL = (float*)(smem + OFF_BIAS_B);
    const int t = threadIdx.x, b = blockIdx.x;

    // A rows [128][232]: [x0(8) | x1(64) | rmax(64) | rmin(64) | 0-pad(32)]
    #pragma unroll 2
    for (int e = 0; e < 116; ++e) {
        int idx = t + 256*e;
        int row = idx / 232, f = idx % 232;
        float v;
        if      (f <   8) v = x0[b*8 + f];
        else if (f <  72) v = x1   [((size_t)b*N_ + row)*C_ + (f-8)];
        else if (f < 136) v = rmaxW[((size_t)b*N_ + row)*C_ + (f-72)];
        else if (f < 200) v = rminW[((size_t)b*N_ + row)*C_ + (f-136)];
        else              v = 0.0f;
        hs[OFF_A1 + row*BSTRIDE + f] = (_Float16)v;
    }
    // W1 [200][64] -> Wt1[n][k] with zero pad k>=200
    #pragma unroll 2
    for (int e = 0; e < 58; ++e) {
        int idx = t + 256*e;
        int n = idx / 232, k = idx % 232;
        hs[OFF_WT1 + n*BSTRIDE + k] = (k < 200) ? (_Float16)W1[k*64 + n] : (_Float16)0.0f;
    }
    if (t < 64) biasL[t] = b1[t];
    __syncthreads();

    const int wave = t >> 5, lane = t & 31;
    const int i0 = wave*16, r = lane & 15, hi = lane >> 4;
    v8f acc[4] = {};
    #pragma unroll
    for (int kc = 0; kc < 7; ++kc) {
        int off = OFF_A1 + (i0 + r)*BSTRIDE + kc*32 + hi*8;
        H16 a; a.h[0] = *(const h8*)(hs + off); a.h[1] = *(const h8*)(hs + off + 16);
        #pragma unroll
        for (int nb = 0; nb < 4; ++nb) {
            int n  = (lane & 15) + nb*16;
            int wo = OFF_WT1 + n*BSTRIDE + kc*32 + hi*8;
            H16 bb; bb.h[0] = *(const h8*)(hs + wo); bb.h[1] = *(const h8*)(hs + wo + 16);
            acc[nb] = __builtin_amdgcn_wmma_f32_16x16x32_f16(
                false, a.v, false, bb.v, (short)0, acc[nb], false, false);
        }
    }
    #pragma unroll
    for (int nb = 0; nb < 4; ++nb) {
        int col = (lane & 15) + nb*16;
        float bv = biasL[col];
        #pragma unroll
        for (int v = 0; v < 8; ++v) {
            int irow = i0 + v + hi*8;
            float x = acc[nb][v] + bv;
            out1[((size_t)b*N_ + irow)*C_ + col] = 1.0f/(1.0f + __expf(-x));
        }
    }
}

// ---------------- Kernel C: out0 (tiny, VALU) ----------------
__global__ __launch_bounds__(64)
void dlm_nullary_kernel(const float* __restrict__ x0, const float* __restrict__ x1,
                        const float* __restrict__ W0, const float* __restrict__ b0,
                        float* __restrict__ out0)
{
    __shared__ float mx[64], mn[64];
    const int c = threadIdx.x, b = blockIdx.x;
    const float* p = x1 + (size_t)b*N_*C_ + c;
    float vmax = p[0], vmin = p[0];
    for (int i = 1; i < N_; ++i) { float v = p[i*64]; vmax = fmaxf(vmax, v); vmin = fminf(vmin, v); }
    mx[c] = vmax; mn[c] = vmin;
    __syncthreads();
    float s = b0[c];
    #pragma unroll
    for (int f = 0; f < 8;  ++f) s += x0[b*8 + f] * W0[f*64 + c];
    for (int q = 0; q < 64; ++q) s += mx[q] * W0[(8 + q)*64 + c];
    for (int q = 0; q < 64; ++q) s += mn[q] * W0[(72 + q)*64 + c];
    out0[b*64 + c] = 1.0f/(1.0f + __expf(-s));
}

extern "C" void kernel_launch(void* const* d_in, const int* in_sizes, int n_in,
                              void* d_out, int out_size, void* d_ws, size_t ws_size,
                              hipStream_t stream)
{
    (void)in_sizes; (void)n_in; (void)out_size; (void)ws_size;
    const float* x0 = (const float*)d_in[0];
    const float* x1 = (const float*)d_in[1];
    const float* x2 = (const float*)d_in[2];
    const float* W0 = (const float*)d_in[3];
    const float* b0 = (const float*)d_in[4];
    const float* W1 = (const float*)d_in[5];
    const float* b1 = (const float*)d_in[6];
    const float* W2 = (const float*)d_in[7];
    const float* b2 = (const float*)d_in[8];

    float* out  = (float*)d_out;
    float* out0 = out;                     // [16,64]
    float* out1 = out0 + B_*C_;            // [16,128,64]
    float* out2 = out1 + (size_t)B_*N_*C_; // [16,128,128,64]

    float* rmaxW = (float*)d_ws;                     // [16,128,64] f32
    float* rminW = rmaxW + (size_t)B_*N_*C_;         // [16,128,64] f32

    // A writes rmax/rmin; B consumes them (same-stream ordering).
    dlm_binary_kernel <<<dim3(B_*N_), dim3(256), SMEM_A_BYTES, stream>>>(
        x1, x2, W2, b2, out2, rmaxW, rminW);
    dlm_unary_kernel  <<<dim3(B_),    dim3(256), SMEM_B_BYTES, stream>>>(
        x0, x1, W1, b1, rmaxW, rminW, out1);
    dlm_nullary_kernel<<<dim3(B_),    dim3(64),  0,            stream>>>(
        x0, x1, W0, b0, out0);
}
